// Transformer_61881888801090
// MI455X (gfx1250) — matrix operand order repaired and checked by reading, not compiled, not run on hardware
//
#include <hip/hip_runtime.h>
#include <hip/hip_bf16.h>
#include <math.h>

// Problem dims (match reference)
#define BB   512
#define SS   64
#define HHD  512
#define NHH  8
#define DKK  64
#define FFD  2048
#define LL   6

typedef __bf16 bf16_t;
typedef __attribute__((ext_vector_type(16))) __bf16 v16bf;
typedef __attribute__((ext_vector_type(8)))  __bf16 v8bf;
typedef __attribute__((ext_vector_type(4)))  __bf16 v4bf;
typedef __attribute__((ext_vector_type(8)))  float  v8f;

// ---------------------------------------------------------------------------
// CDNA5 async global->LDS copy (ASYNCcnt-tracked DMA, no VGPR round trip).
// ---------------------------------------------------------------------------
__device__ __forceinline__ void async_copy_b128(const bf16_t* lds_dst,
                                                const bf16_t* gsrc) {
  unsigned lds_off = (unsigned)(unsigned long long)lds_dst;
  asm volatile("global_load_async_to_lds_b128 %0, %1, off"
               :: "v"(lds_off), "v"(gsrc) : "memory");
}
__device__ __forceinline__ void wait_async_le6() {
  asm volatile("s_wait_asynccnt 0x6" ::: "memory");
}
__device__ __forceinline__ void wait_async_0() {
  asm volatile("s_wait_asynccnt 0x0" ::: "memory");
}

// Concatenate two 8-wide bf16 vectors into the 16-wide WMMA fragment.
__device__ __forceinline__ v16bf frag_cat(v8bf lo, v8bf hi) {
  return __builtin_shufflevector(lo, hi, 0,1,2,3,4,5,6,7,8,9,10,11,12,13,14,15);
}

// A-matrix fragment (16x32 bf16, ISA 7.12.2): lane = {lm = lane&15 -> row,
// lh = lane>>4}.  Elements 0..7 hold K = kofs + 8*lh .. +7, elements 8..15
// hold K = kofs + 8*lh + 16 .. +23.
__device__ __forceinline__ v16bf load_frag_a(const bf16_t* rowptr, int kofs, int lh) {
  v8bf lo = *(const v8bf*)(rowptr + kofs + lh * 8);
  v8bf hi = *(const v8bf*)(rowptr + kofs + lh * 8 + 16);
  return frag_cat(lo, hi);
}

// B-matrix fragment (32x16 bf16): lane {lm -> col, lh}.  Elements 0..7 hold
// K = kofs + 8*lh .. +7, elements 8..15 hold K = kofs + 8*lh + 16 .. +23.
__device__ __forceinline__ v16bf load_frag_b(const bf16_t* colptr, int kofs, int lh) {
  v8bf lo = *(const v8bf*)(colptr + kofs + lh * 8);
  v8bf hi = *(const v8bf*)(colptr + kofs + lh * 8 + 16);
  return frag_cat(lo, hi);
}

__device__ __forceinline__ v8f wmma_bf16(v16bf a, v16bf b, v8f c) {
  return __builtin_amdgcn_wmma_f32_16x16x32_bf16(false, a, false, b, (short)0, c,
                                                 false, false);
}

// Wave32 shuffle reduction (no barriers)
__device__ __forceinline__ float wave_sum(float v) {
#pragma unroll
  for (int off = 16; off > 0; off >>= 1)
    v += __shfl_xor(v, off, 32);
  return v;
}

// ---------------------------------------------------------------------------
// Elementwise f32 -> bf16 (one-time weight conversion; n multiple of 1024)
// ---------------------------------------------------------------------------
__global__ __launch_bounds__(256)
void cvt_f32_bf16(const float* __restrict__ in, bf16_t* __restrict__ out) {
  size_t i = (size_t)blockIdx.x * 256 + threadIdx.x;   // float4 index
  float4 v = ((const float4*)in)[i];
  v4bf t = { (__bf16)v.x, (__bf16)v.y, (__bf16)v.z, (__bf16)v.w };
  ((v4bf*)out)[i] = t;
}

// ---------------------------------------------------------------------------
// GEMM: OUT[m,n] = sum_k A[m,k] * W[n,k] + bias[n]   (A:[M,K] bf16, W:[N,K] bf16)
// 256 thr = 8 waves (2x4 grid), block tile 128x256, wave tile 64x64,
// K stepped by 32.  Double-buffered LDS filled by async global->LDS DMA
// (6 DMAs/wave/tile -> s_wait_asynccnt 0x6 pipelines one tile deep).
// Per K-step per wave: 16 ds_load_b128 -> 16 v_wmma (1:1).
// ---------------------------------------------------------------------------
__global__ __launch_bounds__(256)
void gemm_bf16_wmma(const bf16_t* __restrict__ A, const bf16_t* __restrict__ W,
                    const float* __restrict__ bias,
                    float* __restrict__ OUTF, bf16_t* __restrict__ OUTB,
                    int N, int K) {
  __shared__ bf16_t As[2][128][40];   // 40-elem stride = 80B (16B aligned)
  __shared__ bf16_t Bs[2][256][40];

  const int tid  = threadIdx.x;
  const int m0   = blockIdx.x * 128;
  const int n0   = blockIdx.y * 256;
  const int lane = tid & 31;
  const int wave = tid >> 5;
  const int wm   = wave >> 2;      // 2 wave rows x 4 wave cols
  const int wn   = wave & 3;
  const int lm   = lane & 15;
  const int lh   = lane >> 4;

  // Per-wave per tile: 2 A-chunk DMAs + 4 B-chunk DMAs.
  auto issue_tile = [&](int buf, int k0) {
#pragma unroll
    for (int i = 0; i < 2; ++i) {
      int c = tid + i * 256;               // 512 chunks: 128 rows x 4
      int row = c >> 2, h8 = (c & 3) * 8;
      async_copy_b128(&As[buf][row][h8], A + (size_t)(m0 + row) * K + k0 + h8);
    }
#pragma unroll
    for (int i = 0; i < 4; ++i) {
      int c = tid + i * 256;               // 1024 chunks: 256 rows x 4
      int row = c >> 2, h8 = (c & 3) * 8;
      async_copy_b128(&Bs[buf][row][h8], W + (size_t)(n0 + row) * K + k0 + h8);
    }
  };

  v8f acc[4][4] = {};
  auto compute = [&](int buf) {
    v16bf af[4], bfr[4];
#pragma unroll
    for (int i = 0; i < 4; ++i)
      af[i] = load_frag_a(&As[buf][wm * 64 + i * 16 + lm][0], 0, lh);
#pragma unroll
    for (int j = 0; j < 4; ++j)
      bfr[j] = load_frag_b(&Bs[buf][wn * 64 + j * 16 + lm][0], 0, lh);
#pragma unroll
    for (int i = 0; i < 4; ++i)
#pragma unroll
      for (int j = 0; j < 4; ++j)
        acc[i][j] = wmma_bf16(af[i], bfr[j], acc[i][j]);
  };

  const int nk = K >> 5;
  issue_tile(0, 0);
  for (int kt = 0; kt < nk - 1; ++kt) {
    issue_tile((kt + 1) & 1, (kt + 1) * 32);
    wait_async_le6();                   // tile kt landed; tile kt+1 in flight
    __syncthreads();
    compute(kt & 1);
    __syncthreads();                    // readers done before buf re-fill
  }
  wait_async_0();
  __syncthreads();
  compute((nk - 1) & 1);

  // Epilogue (C layout: row = r + 8*lh within a 16x16 tile, col = lm)
#pragma unroll
  for (int j = 0; j < 4; ++j) {
    int col = n0 + wn * 64 + j * 16 + lm;
    float bv = bias ? bias[col] : 0.0f;
#pragma unroll
    for (int i = 0; i < 4; ++i) {
      int rowb = m0 + wm * 64 + i * 16 + lh * 8;
#pragma unroll
      for (int r = 0; r < 8; ++r) {
        float v = acc[i][j][r] + bv;
        size_t idx = (size_t)(rowb + r) * N + col;
        if (OUTF) OUTF[idx] = v;
        if (OUTB) OUTB[idx] = (__bf16)v;
      }
    }
  }
}

// ---------------------------------------------------------------------------
// Fused SDPA for one (batch, head): S=64, DK=64, bf16 in / bf16 out.
// 128 threads (4 waves); wave w owns query rows [16w, 16w+16).
// Q/K tiles staged by async DMA; V staged transposed (Vt[d][k]) so P*V
// B-fragments stay K-contiguous.  f32 softmax in LDS.
// ---------------------------------------------------------------------------
__global__ __launch_bounds__(128)
void attention_wmma(const bf16_t* __restrict__ Q, const bf16_t* __restrict__ Kp,
                    const bf16_t* __restrict__ Vp, const int* __restrict__ mask,
                    int causal, bf16_t* __restrict__ OUT) {
  __shared__ bf16_t Qs[64][72];
  __shared__ bf16_t Ks[64][72];
  __shared__ bf16_t Vt[64][72];   // Vt[d][k] = V[k][d]
  __shared__ bf16_t Ps[64][72];
  __shared__ float  Sc[64][65];

  const int tid = threadIdx.x;
  const int b = blockIdx.x / NHH;
  const int h = blockIdx.x % NHH;
  const size_t base = ((size_t)b * SS) * HHD + (size_t)h * DKK;

  // Q, K: straight byte copies -> async DMA (512 16B chunks each, 4/thread)
#pragma unroll
  for (int i = 0; i < 4; ++i) {
    int c = tid + i * 128;
    int row = c >> 3, h8 = (c & 7) * 8;
    size_t g = base + (size_t)row * HHD + h8;
    async_copy_b128(&Qs[row][h8], Q + g);
    async_copy_b128(&Ks[row][h8], Kp + g);
  }
  // V: transposed store (1024 4-elem chunks, 8/thread)
#pragma unroll
  for (int i = 0; i < 8; ++i) {
    int c = tid + i * 128;
    int row = c >> 4, h4 = (c & 15) * 4;
    v4bf v = *(const v4bf*)(Vp + base + (size_t)row * HHD + h4);
    Vt[h4 + 0][row] = v[0];
    Vt[h4 + 1][row] = v[1];
    Vt[h4 + 2][row] = v[2];
    Vt[h4 + 3][row] = v[3];
  }
  wait_async_0();
  __syncthreads();

  const int lane = tid & 31, wave = tid >> 5;
  const int lm = lane & 15, lh = lane >> 4;
  const int qb = wave * 16;

  // scores strip [qb, qb+16) x 64
  v8f sacc[4] = {};
#pragma unroll
  for (int kt = 0; kt < 2; ++kt) {
    v16bf a = load_frag_a(&Qs[qb + lm][0], kt * 32, lh);
#pragma unroll
    for (int ct = 0; ct < 4; ++ct) {
      v16bf bfr = load_frag_b(&Ks[ct * 16 + lm][0], kt * 32, lh);
      sacc[ct] = wmma_bf16(a, bfr, sacc[ct]);
    }
  }
#pragma unroll
  for (int ct = 0; ct < 4; ++ct) {
    int col = ct * 16 + lm;
    bool kmask = (mask[b * SS + col] != 0);
#pragma unroll
    for (int r = 0; r < 8; ++r) {
      int row = qb + lh * 8 + r;
      bool keep = kmask && (!causal || (col <= row));
      Sc[row][col] = keep ? sacc[ct][r] * 0.125f : -1e9f;
    }
  }
  __syncthreads();

  // Row softmax in f32, write P as bf16
  if (tid < 64) {
    float m = -3.4e38f;
    for (int k = 0; k < 64; ++k) m = fmaxf(m, Sc[tid][k]);
    float s = 0.0f;
    for (int k = 0; k < 64; ++k) s += expf(Sc[tid][k] - m);
    float inv = 1.0f / s;
    for (int k = 0; k < 64; ++k)
      Ps[tid][k] = (__bf16)(expf(Sc[tid][k] - m) * inv);
  }
  __syncthreads();

  // out strip = P x V
  v8f oacc[4] = {};
#pragma unroll
  for (int kt = 0; kt < 2; ++kt) {
    v16bf a = load_frag_a(&Ps[qb + lm][0], kt * 32, lh);
#pragma unroll
    for (int dt = 0; dt < 4; ++dt) {
      v16bf bfr = load_frag_b(&Vt[dt * 16 + lm][0], kt * 32, lh);
      oacc[dt] = wmma_bf16(a, bfr, oacc[dt]);
    }
  }
#pragma unroll
  for (int dt = 0; dt < 4; ++dt) {
    int d = dt * 16 + lm;
#pragma unroll
    for (int r = 0; r < 8; ++r) {
      int row = qb + lh * 8 + r;
      OUT[base + (size_t)row * HHD + d] = (__bf16)oacc[dt][r];
    }
  }
}

// ---------------------------------------------------------------------------
// OUTF/OUTB[token] = LayerNorm(X[token] + R[token]) * g + b   (block/token)
// Wave32 shuffle reductions + one 8-partial LDS combine per pass.
// ---------------------------------------------------------------------------
__global__ __launch_bounds__(256)
void add_ln_kernel(const float* __restrict__ X, const float* __restrict__ R,
                   const float* __restrict__ g, const float* __restrict__ bb,
                   float* __restrict__ OUTF, bf16_t* __restrict__ OUTB) {
  __shared__ float part[8];
  const int tid = threadIdx.x;
  const int lane = tid & 31, wave = tid >> 5;
  const size_t o = (size_t)blockIdx.x * HHD;

  float v0 = X[o + tid] + R[o + tid];
  float v1 = X[o + tid + 256] + R[o + tid + 256];

  float s = wave_sum(v0 + v1);
  if (lane == 0) part[wave] = s;
  __syncthreads();
  float tot = part[0] + part[1] + part[2] + part[3] +
              part[4] + part[5] + part[6] + part[7];
  float mean = tot * (1.0f / HHD);
  __syncthreads();

  float d0 = v0 - mean, d1 = v1 - mean;
  float q = wave_sum(d0 * d0 + d1 * d1);
  if (lane == 0) part[wave] = q;
  __syncthreads();
  float var = (part[0] + part[1] + part[2] + part[3] +
               part[4] + part[5] + part[6] + part[7]) * (1.0f / HHD);
  float rstd = rsqrtf(var + 1e-5f);

  float r0 = d0 * rstd * g[tid] + bb[tid];
  float r1 = d1 * rstd * g[tid + 256] + bb[tid + 256];
  OUTF[o + tid]       = r0;
  OUTF[o + tid + 256] = r1;
  if (OUTB) {
    OUTB[o + tid]       = (__bf16)r0;
    OUTB[o + tid + 256] = (__bf16)r1;
  }
}

// ---------------------------------------------------------------------------
// x[token] = emb[ids[token]] + sinusoidal PE(token % S); f32 + bf16 outputs
// ---------------------------------------------------------------------------
__global__ __launch_bounds__(256)
void embed_pe_kernel(const int* __restrict__ ids, const float* __restrict__ emb,
                     float* __restrict__ OUTF, bf16_t* __restrict__ OUTB) {
  const int t = blockIdx.x;
  const int s = t % SS;
  const int id = ids[t];
  for (int i = threadIdx.x; i < HHD; i += 256) {
    int i2 = i >> 1;
    float ang = (float)s * expf(-(2.0f * (float)i2 / (float)HHD) * 9.210340372f);
    float pe = (i & 1) ? cosf(ang) : sinf(ang);
    float v = emb[(size_t)id * HHD + i] + pe;
    OUTF[(size_t)t * HHD + i] = v;
    OUTB[(size_t)t * HHD + i] = (__bf16)v;
  }
}

// ---------------------------------------------------------------------------
extern "C" void kernel_launch(void* const* d_in, const int* in_sizes, int n_in,
                              void* d_out, int out_size, void* d_ws, size_t ws_size,
                              hipStream_t stream) {
  (void)in_sizes; (void)n_in; (void)out_size; (void)ws_size;

  const int*   src_ids   = (const int*)  d_in[0];
  const int*   tgt_ids   = (const int*)  d_in[1];
  const int*   src_mask  = (const int*)  d_in[2];
  const int*   tgt_mask  = (const int*)  d_in[3];
  const float* src_emb   = (const float*)d_in[4];
  const float* tgt_emb   = (const float*)d_in[5];
  const float* enc_wq    = (const float*)d_in[6];
  const float* enc_wk    = (const float*)d_in[7];
  const float* enc_wv    = (const float*)d_in[8];
  const float* enc_wo    = (const float*)d_in[9];
  const float* enc_ln_g  = (const float*)d_in[10];
  const float* enc_ln_b  = (const float*)d_in[11];
  const float* enc_fc1_w = (const float*)d_in[12];
  const float* enc_fc1_b = (const float*)d_in[13];
  const float* enc_fc2_w = (const float*)d_in[14];
  const float* enc_fc2_b = (const float*)d_in[15];
  const float* dec_mwq   = (const float*)d_in[16];
  const float* dec_mwk   = (const float*)d_in[17];
  const float* dec_mwv   = (const float*)d_in[18];
  const float* dec_mwo   = (const float*)d_in[19];
  const float* dec_cwq   = (const float*)d_in[20];
  const float* dec_cwo   = (const float*)d_in[21];
  const float* dec_ln_g  = (const float*)d_in[22];
  const float* dec_ln_b  = (const float*)d_in[23];
  const float* dec_fc_w  = (const float*)d_in[24];
  const float* dec_fc_b  = (const float*)d_in[25];

  const int    Mtok = BB * SS;               // 32768 tokens
  const size_t SZ   = (size_t)Mtok * HHD;    // elements per [B,S,H] tensor
  const size_t WHH  = (size_t)HHD * HHD;
  const size_t WL   = (size_t)LL * WHH;      // one [L,H,H] weight stack
  const size_t WF   = (size_t)LL * FFD * HHD;

  // Workspace bump allocator (256B aligned)
  unsigned char* cur = (unsigned char*)d_ws;
  auto alloc_f32 = [&](size_t n) {
    float* p = (float*)cur; cur += (n * 4 + 255) & ~(size_t)255; return p; };
  auto alloc_b16 = [&](size_t n) {
    bf16_t* p = (bf16_t*)cur; cur += (n * 2 + 255) & ~(size_t)255; return p; };

  float*  x   = alloc_f32(SZ);   // encoder stream (f32 master)
  float*  y   = alloc_f32(SZ);   // decoder stream
  float*  T4  = alloc_f32(SZ);   // projection/FFN output (residual branch)
  float*  T5  = alloc_f32(SZ);   // t1 / t2
  bf16_t* xb  = alloc_b16(SZ);   // bf16 shadows
  bf16_t* yb  = alloc_b16(SZ);
  bf16_t* Keb = alloc_b16(SZ);   // last-encoder-layer K (merged head layout)
  bf16_t* Veb = alloc_b16(SZ);
  bf16_t* Qb  = alloc_b16(SZ);   // Q / Qc
  bf16_t* Kdb = alloc_b16(SZ);   // decoder self-attn K
  bf16_t* Vdb = alloc_b16(SZ);   // decoder self-attn V
  bf16_t* T3b = alloc_b16(SZ);   // attention output (merged heads)
  bf16_t* T5b = alloc_b16(SZ);   // t1 / t2 bf16 shadow
  bf16_t* ffb = alloc_b16((size_t)4096 * FFD);  // FFN chunk intermediate

  // One-time weight conversion f32 -> bf16
  auto wcvt = [&](const float* src, size_t n) {
    bf16_t* dst = alloc_b16(n);
    cvt_f32_bf16<<<dim3((unsigned)(n / 1024)), 256, 0, stream>>>(src, dst);
    return dst;
  };
  bf16_t* enc_wq_b  = wcvt(enc_wq,  WL);
  bf16_t* enc_wk_b  = wcvt(enc_wk,  WL);
  bf16_t* enc_wv_b  = wcvt(enc_wv,  WL);
  bf16_t* enc_wo_b  = wcvt(enc_wo,  WL);
  bf16_t* enc_f1_b  = wcvt(enc_fc1_w, WF);
  bf16_t* enc_f2_b  = wcvt(enc_fc2_w, WF);
  bf16_t* dec_mwq_b = wcvt(dec_mwq, WL);
  bf16_t* dec_mwk_b = wcvt(dec_mwk, WL);
  bf16_t* dec_mwv_b = wcvt(dec_mwv, WL);
  bf16_t* dec_mwo_b = wcvt(dec_mwo, WL);
  bf16_t* dec_cwq_b = wcvt(dec_cwq, WL);
  bf16_t* dec_cwo_b = wcvt(dec_cwo, WL);
  bf16_t* dec_fcw_b = wcvt(dec_fc_w, WL);

  auto gemm = [&](const bf16_t* A, const bf16_t* W, const float* bias,
                  float* OF, bf16_t* OB, int M, int N, int K) {
    dim3 grid(M / 128, N / 256);
    gemm_bf16_wmma<<<grid, 256, 0, stream>>>(A, W, bias, OF, OB, N, K);
  };

  embed_pe_kernel<<<Mtok, 256, 0, stream>>>(src_ids, src_emb, x, xb);
  embed_pe_kernel<<<Mtok, 256, 0, stream>>>(tgt_ids, tgt_emb, y, yb);

  // -------------------- Encoder --------------------
  for (int l = 0; l < LL; ++l) {
    gemm(xb, enc_wq_b + (size_t)l * WHH, nullptr, nullptr, Qb,  Mtok, HHD, HHD);
    gemm(xb, enc_wk_b + (size_t)l * WHH, nullptr, nullptr, Keb, Mtok, HHD, HHD);
    gemm(xb, enc_wv_b + (size_t)l * WHH, nullptr, nullptr, Veb, Mtok, HHD, HHD);
    attention_wmma<<<BB * NHH, 128, 0, stream>>>(Qb, Keb, Veb, src_mask, 0, T3b);
    gemm(T3b, enc_wo_b + (size_t)l * WHH, nullptr, T4, nullptr, Mtok, HHD, HHD);
    add_ln_kernel<<<Mtok, 256, 0, stream>>>(x, T4, enc_ln_g + l * HHD,
                                            enc_ln_b + l * HHD, x, xb);
    // FFN chunked over M (8 x 4096 rows) to bound workspace
    for (int c = 0; c < 8; ++c) {
      gemm(xb + (size_t)c * 4096 * HHD, enc_f1_b + (size_t)l * FFD * HHD,
           enc_fc1_b + l * FFD, nullptr, ffb, 4096, FFD, HHD);
      gemm(ffb, enc_f2_b + (size_t)l * HHD * FFD, enc_fc2_b + l * HHD,
           T4 + (size_t)c * 4096 * HHD, nullptr, 4096, HHD, FFD);
    }
    add_ln_kernel<<<Mtok, 256, 0, stream>>>(x, T4, enc_ln_g + l * HHD,
                                            enc_ln_b + l * HHD, x, xb);
  }
  // Keb/Veb now hold the LAST encoder layer's K,V (feed cross-attention).

  // -------------------- Decoder --------------------
  for (int l = 0; l < LL; ++l) {
    // masked self-attention
    gemm(yb, dec_mwq_b + (size_t)l * WHH, nullptr, nullptr, Qb,  Mtok, HHD, HHD);
    gemm(yb, dec_mwk_b + (size_t)l * WHH, nullptr, nullptr, Kdb, Mtok, HHD, HHD);
    gemm(yb, dec_mwv_b + (size_t)l * WHH, nullptr, nullptr, Vdb, Mtok, HHD, HHD);
    attention_wmma<<<BB * NHH, 128, 0, stream>>>(Qb, Kdb, Vdb, tgt_mask, 1, T3b);
    gemm(T3b, dec_mwo_b + (size_t)l * WHH, nullptr, T4, nullptr, Mtok, HHD, HHD);
    add_ln_kernel<<<Mtok, 256, 0, stream>>>(y, T4, dec_ln_g + l * HHD,
                                            dec_ln_b + l * HHD, T5, T5b);  // t1
    // cross-attention (reference applies the TARGET causal mask — faithful)
    gemm(T5b, dec_cwq_b + (size_t)l * WHH, nullptr, nullptr, Qb, Mtok, HHD, HHD);
    attention_wmma<<<BB * NHH, 128, 0, stream>>>(Qb, Keb, Veb, tgt_mask, 1, T3b);
    gemm(T3b, dec_cwo_b + (size_t)l * WHH, nullptr, T4, nullptr, Mtok, HHD, HHD);
    add_ln_kernel<<<Mtok, 256, 0, stream>>>(T5, T4, dec_ln_g + l * HHD,
                                            dec_ln_b + l * HHD, T5, T5b);  // t2
    // decoder feed-forward (single H x H linear)
    gemm(T5b, dec_fcw_b + (size_t)l * WHH, dec_fc_b + l * HHD, T4, nullptr,
         Mtok, HHD, HHD);
    if (l == LL - 1) {
      add_ln_kernel<<<Mtok, 256, 0, stream>>>(T5, T4, dec_ln_g + l * HHD,
                                              dec_ln_b + l * HHD,
                                              (float*)d_out, nullptr);
    } else {
      add_ln_kernel<<<Mtok, 256, 0, stream>>>(T5, T4, dec_ln_g + l * HHD,
                                              dec_ln_b + l * HHD, y, yb);
    }
  }
}
